// UnionBoxesAndFeats_3882650435976
// MI455X (gfx1250) — matrix operand order repaired and checked by reading, not compiled, not run on hardware
//
#include <hip/hip_runtime.h>

typedef __attribute__((ext_vector_type(16))) _Float16 v16h;
typedef __attribute__((ext_vector_type(8)))  float    v8f;

#define NP 4096

// ---------------- workspace layout (bytes) ----------------
static constexpr size_t OFF_W1P  = 0;                          // 128*128*2  (pre-swizzled A frags)
static constexpr size_t OFF_W2P  = OFF_W1P  + 128*128*2;       // 256*1152*2 (pre-swizzled A frags)
static constexpr size_t OFF_ST1  = OFF_W2P  + 256*1152*2;      // 256*4
static constexpr size_t OFF_ST2  = OFF_ST1  + 256*4;           // 512*4
static constexpr size_t OFF_PB1  = OFF_ST2  + 512*4;           // 2*128*NP*4 = 4 MB
static constexpr size_t OFF_PB2  = OFF_PB1  + (size_t)2*128*NP*4;   // 2*256*NP*4 = 8 MB
static constexpr size_t OFF_RECT = OFF_PB2  + (size_t)2*256*NP*4;   // NP*1458*2
static constexpr size_t OFF_H1   = OFF_RECT + (size_t)NP*1458*2;    // NP*128*196*2
static constexpr size_t OFF_H1P  = OFF_H1   + (size_t)NP*128*196*2; // NP*128*49*2
static constexpr size_t OFF_Y2R  = OFF_H1P  + (size_t)NP*128*49*2;  // NP*256*49*2

__device__ __forceinline__ int kpair_of(int j, int half) {
  return 16 * half + j;   // contiguous k order, matches B fragment loads
}

// ---------------- weight f32 -> f16, pre-swizzled into per-fragment order ----------------
__global__ __launch_bounds__(256) void k_cvt_w1(const float* __restrict__ W1,
                                                _Float16* __restrict__ W1p) {
  const int i = blockIdx.x * 256 + threadIdx.x;     // 16384
  if (i >= 128 * 128) return;
  const int j = i & 15, lane = (i >> 4) & 31, r = i >> 9;
  const int ks = r & 3, mt = r >> 2;
  const int k = ks * 32 + kpair_of(j, lane >> 4);
  const int m = mt * 16 + (lane & 15);
  W1p[i] = (_Float16)(k < 98 ? W1[m * 98 + k] : 0.0f);
}

__global__ __launch_bounds__(256) void k_cvt_w2(const float* __restrict__ W2,
                                                _Float16* __restrict__ W2p) {
  const int i = blockIdx.x * 256 + threadIdx.x;     // 294912
  if (i >= 256 * 1152) return;
  const int j = i & 15, lane = (i >> 4) & 31, r = i >> 9;
  const int ks = r % 36, mt = r / 36;
  const int k = ks * 32 + kpair_of(j, lane >> 4);
  const int m = mt * 16 + (lane & 15);
  W2p[i] = (_Float16)W2[m * 1152 + k];
}

// ---------------- rasterize pair boxes: rects - 0.5 (f16) ----------------
__global__ __launch_bounds__(256) void k_rects(const float* __restrict__ rois,
                                               const int* __restrict__ uinds,
                                               _Float16* __restrict__ rects) {
  const int p = blockIdx.x;
  __shared__ float sx1[2], sx2[2], sy1[2], sy2[2];
  if (threadIdx.x == 0) {
    const int i0 = uinds[2 * p], i1 = uinds[2 * p + 1];
    float bx[2][4];
#pragma unroll
    for (int j = 0; j < 4; ++j) { bx[0][j] = rois[i0 * 5 + 1 + j]; bx[1][j] = rois[i1 * 5 + 1 + j]; }
    const float ux1 = fminf(bx[0][0], bx[1][0]);
    const float uy1 = fminf(bx[0][1], bx[1][1]);
    const float ux2 = fmaxf(bx[0][2], bx[1][2]);
    const float uy2 = fmaxf(bx[0][3], bx[1][3]);
    const float w = ux2 - ux1, h = uy2 - uy1;
#pragma unroll
    for (int k = 0; k < 2; ++k) {
      sx1[k] = (bx[k][0] - ux1) / w * 27.0f;  sx2[k] = (bx[k][2] - ux1) / w * 27.0f;
      sy1[k] = (bx[k][1] - uy1) / h * 27.0f;  sy2[k] = (bx[k][3] - uy1) / h * 27.0f;
    }
  }
  __syncthreads();
  for (int idx = threadIdx.x; idx < 1458; idx += 256) {
    const int k = idx / 729, rem = idx % 729, y = rem / 27, x = rem % 27;
    const float cx = fminf(fmaxf(fminf((float)x + 1.0f, sx2[k]) - fmaxf((float)x, sx1[k]), 0.0f), 1.0f);
    const float cy = fminf(fmaxf(fminf((float)y + 1.0f, sy2[k]) - fmaxf((float)y, sy1[k]), 0.0f), 1.0f);
    rects[(size_t)p * 1458 + idx] = (_Float16)(cy * cx - 0.5f);
  }
}

// ---------------- RoIAlign of union boxes -> d_out ----------------
__global__ __launch_bounds__(256) void k_roialign(const float* __restrict__ fmap,
                                                  const float* __restrict__ rois,
                                                  const int* __restrict__ uinds,
                                                  float* __restrict__ out) {
  const int p = blockIdx.x;
  __shared__ float sxw[7], syw[7];
  __shared__ int sx0[7], sx1i[7], sy0[7], sy1i[7];
  const int t = threadIdx.x;
  if (t < 7) {
    const int i0 = uinds[2 * p], i1 = uinds[2 * p + 1];
    const float x1 = fminf(rois[i0 * 5 + 1], rois[i1 * 5 + 1]);
    const float y1 = fminf(rois[i0 * 5 + 2], rois[i1 * 5 + 2]);
    const float x2 = fmaxf(rois[i0 * 5 + 3], rois[i1 * 5 + 3]);
    const float y2 = fmaxf(rois[i0 * 5 + 4], rois[i1 * 5 + 4]);
    const float s = 1.0f / 16.0f;
    const float tt = (float)t * (1.0f / 6.0f);
    const float xs = fminf(fmaxf((x1 * (1.0f - tt) + x2 * tt) * s, 0.0f), 36.0f);
    const float ys = fminf(fmaxf((y1 * (1.0f - tt) + y2 * tt) * s, 0.0f), 36.0f);
    const float x0 = floorf(xs), y0 = floorf(ys);
    sx0[t] = (int)x0; sx1i[t] = min((int)x0 + 1, 36); sxw[t] = xs - x0;
    sy0[t] = (int)y0; sy1i[t] = min((int)y0 + 1, 36); syw[t] = ys - y0;
  }
  __syncthreads();
  for (int idx = t; idx < 256 * 49; idx += 256) {
    const int ch = idx / 49, pos = idx % 49, gy = pos / 7, gx = pos % 7;
    const float* f = fmap + (size_t)ch * 1369;
    const float wx = sxw[gx], wy = syw[gy];
    const int x0 = sx0[gx], x1 = sx1i[gx], y0 = sy0[gy], y1 = sy1i[gy];
    const float v00 = f[y0 * 37 + x0], v01 = f[y0 * 37 + x1];
    const float v10 = f[y1 * 37 + x0], v11 = f[y1 * 37 + x1];
    out[(size_t)p * (256 * 49) + idx] =
        v00 * (1.0f - wy) * (1.0f - wx) + v01 * (1.0f - wy) * wx +
        v10 * wy * (1.0f - wx) + v11 * wy * wx;
  }
}

// ---------------- conv1 (WMMA f16) + fused deterministic BN-stat partials ----------------
__global__ __launch_bounds__(256) void k_conv1(const _Float16* __restrict__ W1p,   // swizzled A
                                               const _Float16* __restrict__ rects, // [NP][1458]
                                               const float* __restrict__ b1,
                                               _Float16* __restrict__ h1,          // [NP][128][196]
                                               float* __restrict__ part) {         // [2][128][NP]
  __shared__ _Float16 spad[3138];                 // [2][33][33] zero-padded + zero tail
  __shared__ alignas(32) _Float16 col[208 * 128]; // im2col [n][k] = 53 KB
  __shared__ int ktab[128];
  __shared__ int dtab[208];
  const int tid = threadIdx.x;
  const int p = blockIdx.x;

  for (int i = tid; i < 3138; i += 256) spad[i] = (_Float16)0.0f;
  if (tid < 128) {
    int e = 2178;                                  // zero-tail slot for padded K
    if (tid < 98) {
      const int c = tid / 49, rem = tid % 49, ky = rem / 7, kx = rem % 7;
      e = c * 1089 + ky * 33 + kx;
    }
    ktab[tid] = e;
  }
  if (tid < 208) {
    const int oy = tid / 14, ox = tid % 14;
    dtab[tid] = (tid < 196) ? (66 * oy + 2 * ox) : -1;
  }
  __syncthreads();
  for (int i = tid; i < 1458; i += 256) {          // fill interior (pad offset +3)
    const int c = i / 729, rem = i % 729, y = rem / 27, x = rem % 27;
    spad[c * 1089 + (y + 3) * 33 + (x + 3)] = rects[(size_t)p * 1458 + i];
  }
  __syncthreads();
  for (int j = 0; j < 104; ++j) {                  // 26624 entries, branch-free
    const int idx = tid + 256 * j;
    const int n = idx >> 7, k = idx & 127;
    const int d = dtab[n];
    const int off = (d >= 0) ? (ktab[k] + d) : 0;  // spad[0] is guaranteed zero
    col[idx] = spad[off];
  }
  __syncthreads();

  const int lane = tid & 31, wv = tid >> 5;
  const int half = lane >> 4, lm = lane & 15;
  const int m0 = wv * 16;                          // each wave owns one M tile

  v16h afr[4];                                     // one 32B contiguous load per fragment
#pragma unroll
  for (int ks = 0; ks < 4; ++ks)
    afr[ks] = *(const v16h*)(W1p + ((size_t)(wv * 4 + ks) * 32 + lane) * 16);

  float ssum[8] = {}, ssq[8] = {};
  for (int nt = 0; nt < 13; ++nt) {
    const int n = nt * 16 + lm;
    const _Float16* brow = &col[(size_t)(nt * 16 + lm) * 128 + half * 16];
    v8f acc = {};
#pragma unroll
    for (int ks = 0; ks < 4; ++ks) {
      const v16h b = *(const v16h*)(brow + ks * 32);   // 2x ds_load_b128
      acc = __builtin_amdgcn_wmma_f32_16x16x32_f16(false, afr[ks], false, b,
                                                   (short)0, acc, false, false);
    }
    const bool ncol = (n < 196);
#pragma unroll
    for (int r = 0; r < 8; ++r) {                  // D row = r + 8*half
      const int ch = m0 + half * 8 + r;
      float v = acc[r] + b1[ch];
      v = fmaxf(v, 0.0f);
      const float vs = ncol ? v : 0.0f;
      ssum[r] += vs;
      ssq[r] += vs * vs;
      if (ncol) h1[((size_t)p * 128 + ch) * 196 + n] = (_Float16)v;
    }
  }
  // deterministic cross-lane reduction within each 16-lane half-group
#pragma unroll
  for (int r = 0; r < 8; ++r) {
#pragma unroll
    for (int m = 8; m >= 1; m >>= 1) {
      ssum[r] += __shfl_xor(ssum[r], m, 16);
      ssq[r]  += __shfl_xor(ssq[r], m, 16);
    }
  }
  if (lm == 0) {
#pragma unroll
    for (int r = 0; r < 8; ++r) {
      const int ch = m0 + half * 8 + r;
      part[(size_t)ch * NP + p] = ssum[r];
      part[(size_t)(128 + ch) * NP + p] = ssq[r];
    }
  }
}

// ---------------- conv2 (WMMA f16) + fused deterministic BN-stat partials ----------------
__global__ __launch_bounds__(256) void k_conv2(const _Float16* __restrict__ W2p, // swizzled A
                                               const _Float16* __restrict__ h1p, // [NP][128][49]
                                               const float* __restrict__ b2,
                                               _Float16* __restrict__ y2r,       // [NP][256][49]
                                               float* __restrict__ part) {       // [2][256][NP]
  __shared__ _Float16 spad[128 * 81];              // [128][9][9] zero-padded (pad=1)
  __shared__ alignas(32) _Float16 chunk[64 * 128]; // im2col k-chunk [n][k] = 16 KB
  __shared__ int ktab[1152];
  __shared__ int dtab[64];
  const int tid = threadIdx.x;
  const int p = blockIdx.x;

  for (int i = tid; i < 128 * 81; i += 256) spad[i] = (_Float16)0.0f;
  for (int k = tid; k < 1152; k += 256) {
    const int c = k / 9, rem = k % 9, ky = rem / 3, kx = rem % 3;
    ktab[k] = c * 81 + ky * 9 + kx;
  }
  if (tid < 64) dtab[tid] = (tid < 49) ? ((tid / 7) * 9 + (tid % 7)) : -1;
  __syncthreads();
  for (int i = tid; i < 128 * 49; i += 256) {      // fill interior (pad offset +1)
    const int c = i / 49, pos = i % 49, y = pos / 7, x = pos % 7;
    spad[c * 81 + (y + 1) * 9 + (x + 1)] = h1p[(size_t)p * (128 * 49) + i];
  }

  const int lane = tid & 31, wv = tid >> 5;
  const int half = lane >> 4, lm = lane & 15;

  v8f acc[2][4] = {};
  for (int kc = 0; kc < 9; ++kc) {                 // K = 1152 in chunks of 128
    __syncthreads();
    for (int j = 0; j < 32; ++j) {                 // build 8192-entry chunk, branch-free
      const int idx = tid + 256 * j;
      const int n = idx >> 7, kk = idx & 127;
      const int d = dtab[n];
      const int off = (d >= 0) ? (ktab[kc * 128 + kk] + d) : 0;  // spad[0] is zero border
      chunk[idx] = spad[off];
    }
    __syncthreads();
#pragma unroll
    for (int ks4 = 0; ks4 < 4; ++ks4) {
      v16h bfr[4];
#pragma unroll
      for (int nt = 0; nt < 4; ++nt)
        bfr[nt] = *(const v16h*)(&chunk[(size_t)(nt * 16 + lm) * 128 + ks4 * 32 + half * 16]);
#pragma unroll
      for (int mi = 0; mi < 2; ++mi) {
        const int mt = wv + 8 * mi;
        const v16h a = *(const v16h*)(W2p + ((size_t)(mt * 36 + kc * 4 + ks4) * 32 + lane) * 16);
#pragma unroll
        for (int nt = 0; nt < 4; ++nt)
          acc[mi][nt] = __builtin_amdgcn_wmma_f32_16x16x32_f16(false, a, false, bfr[nt],
                                                               (short)0, acc[mi][nt], false, false);
      }
    }
  }

  float ssum[2][8] = {}, ssq[2][8] = {};
#pragma unroll
  for (int mi = 0; mi < 2; ++mi) {
    const int m0 = (wv + 8 * mi) * 16;
#pragma unroll
    for (int nt = 0; nt < 4; ++nt) {
      const int n = nt * 16 + lm;
      const bool ncol = (n < 49);
#pragma unroll
      for (int r = 0; r < 8; ++r) {
        const int ch = m0 + half * 8 + r;
        float v = acc[mi][nt][r] + b2[ch];
        v = fmaxf(v, 0.0f);
        const float vs = ncol ? v : 0.0f;
        ssum[mi][r] += vs;
        ssq[mi][r] += vs * vs;
        if (ncol) y2r[((size_t)p * 256 + ch) * 49 + n] = (_Float16)v;
      }
    }
  }
#pragma unroll
  for (int mi = 0; mi < 2; ++mi)
#pragma unroll
    for (int r = 0; r < 8; ++r) {
#pragma unroll
      for (int m = 8; m >= 1; m >>= 1) {
        ssum[mi][r] += __shfl_xor(ssum[mi][r], m, 16);
        ssq[mi][r]  += __shfl_xor(ssq[mi][r], m, 16);
      }
    }
  if (lm == 0) {
#pragma unroll
    for (int mi = 0; mi < 2; ++mi)
#pragma unroll
      for (int r = 0; r < 8; ++r) {
        const int ch = (wv + 8 * mi) * 16 + half * 8 + r;
        part[(size_t)ch * NP + p] = ssum[mi][r];
        part[(size_t)(256 + ch) * NP + p] = ssq[mi][r];
      }
  }
}

// ---------------- deterministic reduction of per-block partials ----------------
template <int C>
__global__ __launch_bounds__(256) void k_red(const float* __restrict__ part,  // [2][C][NP]
                                             float* __restrict__ stats) {     // [2][C]
  const int ch = blockIdx.x;
  const int t = threadIdx.x;
  __shared__ float s1[256], s2[256];
  float a = 0.0f, b = 0.0f;
  for (int i = t; i < NP; i += 256) {
    a += part[(size_t)ch * NP + i];
    b += part[(size_t)(C + ch) * NP + i];
  }
  s1[t] = a; s2[t] = b;
  __syncthreads();
  for (int st = 128; st > 0; st >>= 1) {
    if (t < st) { s1[t] += s1[t + st]; s2[t] += s2[t + st]; }
    __syncthreads();
  }
  if (t == 0) { stats[ch] = s1[0]; stats[C + ch] = s2[0]; }
}

// ---------------- BN(normalize) + maxpool 3x3 s2 p1 (14x14 -> 7x7) ----------------
__global__ __launch_bounds__(256) void k_bnpool(const _Float16* __restrict__ h1,
                                                const float* __restrict__ st,
                                                const float* __restrict__ g,
                                                const float* __restrict__ be,
                                                _Float16* __restrict__ h1p) {
  const int p = blockIdx.x;
  const float invN = 1.0f / ((float)NP * 196.0f);
  for (int idx = threadIdx.x; idx < 128 * 49; idx += 256) {
    const int ch = idx / 49, pos = idx % 49, oy = pos / 7, ox = pos % 7;
    const float mean = st[ch] * invN;
    const float var = st[128 + ch] * invN - mean * mean;
    const float sc = g[ch] * rsqrtf(var + 1e-5f);
    const float sh = be[ch] - mean * sc;
    const _Float16* src = h1 + ((size_t)p * 128 + ch) * 196;
    float m = -3.4e38f;
#pragma unroll
    for (int dy = 0; dy < 3; ++dy) {
      const int y = 2 * oy - 1 + dy;
      if ((unsigned)y >= 14u) continue;
#pragma unroll
      for (int dx = 0; dx < 3; ++dx) {
        const int x = 2 * ox - 1 + dx;
        if ((unsigned)x >= 14u) continue;
        m = fmaxf(m, sc * (float)src[y * 14 + x] + sh);
      }
    }
    h1p[(size_t)p * (128 * 49) + idx] = (_Float16)m;
  }
}

// ---------------- final BN + add into d_out ----------------
__global__ __launch_bounds__(256) void k_final(const _Float16* __restrict__ y2r,
                                               const float* __restrict__ st,
                                               const float* __restrict__ g,
                                               const float* __restrict__ be,
                                               float* __restrict__ out) {
  const int p = blockIdx.x;
  const float invN = 1.0f / ((float)NP * 49.0f);
  for (int idx = threadIdx.x; idx < 256 * 49; idx += 256) {
    const int ch = idx / 49;
    const float mean = st[ch] * invN;
    const float var = st[256 + ch] * invN - mean * mean;
    const float sc = g[ch] * rsqrtf(var + 1e-5f);
    const float sh = be[ch] - mean * sc;
    const size_t o = (size_t)p * (256 * 49) + idx;
    out[o] += sc * (float)y2r[o] + sh;
  }
}

extern "C" void kernel_launch(void* const* d_in, const int* in_sizes, int n_in,
                              void* d_out, int out_size, void* d_ws, size_t ws_size,
                              hipStream_t stream) {
  const float* fmap  = (const float*)d_in[0];
  const float* rois  = (const float*)d_in[1];
  const int*   uinds = (const int*)d_in[2];
  const float* W1    = (const float*)d_in[3];
  const float* b1    = (const float*)d_in[4];
  const float* g1    = (const float*)d_in[5];
  const float* be1   = (const float*)d_in[6];
  const float* W2    = (const float*)d_in[7];
  const float* b2    = (const float*)d_in[8];
  const float* g2    = (const float*)d_in[9];
  const float* be2   = (const float*)d_in[10];
  float* out = (float*)d_out;
  char* ws = (char*)d_ws;

  _Float16* W1p   = (_Float16*)(ws + OFF_W1P);
  _Float16* W2p   = (_Float16*)(ws + OFF_W2P);
  float*    st1   = (float*)(ws + OFF_ST1);
  float*    st2   = (float*)(ws + OFF_ST2);
  float*    pb1   = (float*)(ws + OFF_PB1);
  float*    pb2   = (float*)(ws + OFF_PB2);
  _Float16* rects = (_Float16*)(ws + OFF_RECT);
  _Float16* h1    = (_Float16*)(ws + OFF_H1);
  _Float16* h1p   = (_Float16*)(ws + OFF_H1P);
  _Float16* y2r   = (_Float16*)(ws + OFF_Y2R);

  k_cvt_w1<<<64, 256, 0, stream>>>(W1, W1p);
  k_cvt_w2<<<1152, 256, 0, stream>>>(W2, W2p);
  k_rects<<<NP, 256, 0, stream>>>(rois, uinds, rects);
  k_roialign<<<NP, 256, 0, stream>>>(fmap, rois, uinds, out);
  k_conv1<<<NP, 256, 0, stream>>>(W1p, rects, b1, h1, pb1);
  k_red<128><<<128, 256, 0, stream>>>(pb1, st1);
  k_bnpool<<<NP, 256, 0, stream>>>(h1, st1, g1, be1, h1p);
  k_conv2<<<NP, 256, 0, stream>>>(W2p, h1p, b2, y2r, pb2);
  k_red<256><<<256, 256, 0, stream>>>(pb2, st2);
  k_final<<<NP, 256, 0, stream>>>(y2r, st2, g2, be2, out);
}
